// Decoder_40089224741460
// MI455X (gfx1250) — compile-verified, matrix operand order checked by reading, not run on hardware
//
#include <hip/hip_runtime.h>
#include <math.h>

typedef __attribute__((ext_vector_type(16))) _Float16 v16h;
typedef __attribute__((ext_vector_type(8)))  float    v8f;
typedef _Float16 h16;

#define B_    16
#define LATD  512
#define HW    4096
#define NF_   32
#define CWCH  24

// ---- CDNA5 WMMA 16-bit tile lane/element mapping ----
__device__ __forceinline__ void pack_pos(int q, int k, int& lane, int& j) {
    int hi = (k >> 3) & 1;
    int kk = k - 8 * hi;          // 0..7 or 16..23
    j = (kk < 8) ? kk : kk - 8;
    lane = q + 16 * hi;
}

// packed tile fetch: one v16h per lane (2x global_load_b128 / ds_load_b128)
__device__ __forceinline__ v16h load_tile(const h16* __restrict__ buf, size_t tile) {
    return *(const v16h*)(buf + tile * 512 + (size_t)(threadIdx.x & 31) * 16);
}

// B tile (32x16) scalar gather from row-major [K][ld] f32 (one-time paths only)
__device__ __forceinline__ v16h load_bm_f32(const float* __restrict__ Bs, size_t ld,
                                            int k0, int n0, int kmax) {
    int l = threadIdx.x & 31, n = l & 15, hi = l >> 4;
    v16h r;
#pragma unroll
    for (int j = 0; j < 16; ++j) {
        int v = j >> 1, e = j & 1;
        int k = k0 + 2 * v + ((v >= 4) ? 8 : 0) + hi * 8 + e;
        float x = (k < kmax) ? Bs[(size_t)k * ld + n0 + n] : 0.0f;
        r[j] = (h16)x;
    }
    return r;
}

__device__ __forceinline__ v8f WMMA(v16h a, v16h b, v8f c) {
    return __builtin_amdgcn_wmma_f32_16x16x32_f16(false, a, false, b, (short)0, c, false, false);
}

// ================= one-time pack kernels =================
__global__ void k_pack_lat(const float* __restrict__ lat, h16* __restrict__ latA) {
    int e = blockIdx.x * 256 + threadIdx.x;        // 16*512
    int m = e / LATD, k = e % LATD;
    int lane, j; pack_pos(m, k & 31, lane, j);
    latA[((size_t)(k >> 5) * 32 + lane) * 16 + j] = (h16)lat[e];
}

__global__ void k_pack_wb(const float* __restrict__ wb, h16* __restrict__ wbB) {
    int e = blockIdx.x * 256 + threadIdx.x;        // 512*4096
    int kch = e / HW, p = e % HW;
    int lane, j; pack_pos(p & 15, kch & 31, lane, j);
    wbB[(((size_t)(kch >> 5) * 256 + (p >> 4)) * 32 + lane) * 16 + j] = (h16)wb[e];
}

__global__ void k_pack_cw(const float* __restrict__ cw, h16* __restrict__ cwB) {
    int e = blockIdx.x * 256 + threadIdx.x;        // 4*32*4096
    int c = e / (32 * HW);
    int rem = e % (32 * HW);
    int kch = rem / HW, p = rem % HW;
    float v = (kch < CWCH) ? cw[((size_t)c * CWCH + kch) * HW + p] : 0.0f;
    int lane, j; pack_pos(p & 15, kch, lane, j);
    cwB[(((size_t)c * 256 + (p >> 4)) * 32 + lane) * 16 + j] = (h16)v;
}

// ================= dynamic weight generation =================
__global__ void k_wd_gemm_rm(const h16* __restrict__ latA, const float* __restrict__ w,
                             h16* __restrict__ wd, int N) {
    int n0 = blockIdx.x * 16;
    int l = threadIdx.x & 31, n = l & 15, hi = l >> 4;
    v8f acc = {};
    for (int kb = 0; kb < LATD / 32; ++kb) {
        v16h a  = load_tile(latA, (size_t)kb);
        v16h bb = load_bm_f32(w, (size_t)N, kb * 32, n0, LATD);
        acc = WMMA(a, bb, acc);
    }
#pragma unroll
    for (int r = 0; r < 8; ++r) {
        int b = r + hi * 8;
        wd[(size_t)b * N + n0 + n] = (h16)acc[r];
    }
}

__global__ void k_wd_gemm_pk(const h16* __restrict__ latA, const float* __restrict__ w,
                             h16* __restrict__ wdA, int O, int I) {
    int N = O * I;
    int n0 = blockIdx.x * 16;
    int l = threadIdx.x & 31, n = l & 15, hi = l >> 4;
    v8f acc = {};
    for (int kb = 0; kb < LATD / 32; ++kb) {
        v16h a  = load_tile(latA, (size_t)kb);
        v16h bb = load_bm_f32(w, (size_t)N, kb * 32, n0, LATD);
        acc = WMMA(a, bb, acc);
    }
    int MT = O >> 4, KB = I >> 5;
    int ncol = n0 + n;
    int o = ncol / I, i = ncol % I;
    int lane2, j2; pack_pos(o & 15, i & 31, lane2, j2);
    size_t cpart = ((size_t)((o >> 4) * KB + (i >> 5)) * 32 + lane2) * 16 + j2;
    size_t bstride = (size_t)MT * KB * 512;
#pragma unroll
    for (int r = 0; r < 8; ++r) {
        int b = r + hi * 8;
        wdA[(size_t)b * bstride + cpart] = (h16)acc[r];
    }
}

// dynamic biases via WMMA: bd(16xO f32) = lat @ wb, column-clamped for O%16 != 0
__global__ void k_bias_gemm(const h16* __restrict__ latA, const float* __restrict__ wb,
                            float* __restrict__ bd, int O) {
    int n0 = blockIdx.x * 16;
    int l = threadIdx.x & 31, n = l & 15, hi = l >> 4;
    int col = n0 + n;
    int colc = (col < O) ? col : O - 1;
    v8f acc = {};
    for (int kb = 0; kb < LATD / 32; ++kb) {
        v16h a = load_tile(latA, (size_t)kb);
        v16h bbv;
#pragma unroll
        for (int j = 0; j < 16; ++j) {
            int v = j >> 1, e = j & 1;
            int k = kb * 32 + 2 * v + ((v >= 4) ? 8 : 0) + hi * 8 + e;
            bbv[j] = (h16)wb[(size_t)k * O + colc];
        }
        acc = WMMA(a, bbv, acc);
    }
    if (col < O) {
#pragma unroll
        for (int r = 0; r < 8; ++r) bd[(size_t)(r + hi * 8) * O + col] = acc[r];
    }
}

// repack w2c columns 24..87 into aligned packed A (64x64, MT=4, KB=2)
__global__ void k_pack_w2cA2(const h16* __restrict__ w2c_rm, h16* __restrict__ w2cA2) {
    int e = blockIdx.x * 256 + threadIdx.x;        // 16*64*64
    int b = e / 4096, rem = e % 4096;
    int o = rem >> 6, k = rem & 63;
    h16 v = w2c_rm[(size_t)b * 5632 + o * 88 + 24 + k];
    int lane, j; pack_pos(o & 15, k & 31, lane, j);
    w2cA2[(((size_t)(b * 4 + (o >> 4)) * 2 + (k >> 5)) * 32 + lane) * 16 + j] = v;
}

__global__ void k_seed(const float* __restrict__ seed, float* __restrict__ emb0) {
    int t = blockIdx.x * blockDim.x + threadIdx.x;
    emb0[t] = seed[t & (NF_ * HW - 1)];
}

// ---- surface GEMM + complex wave synthesis; writes wave_b packed-B (KB=2) ----
__global__ void k_surface(const h16* __restrict__ btA, const float* __restrict__ b_bt,
                          const h16* __restrict__ wbB, h16* __restrict__ waveB) {
    int b = blockIdx.y, nt = blockIdx.x;
    int mt = threadIdx.x >> 5;
    int l = threadIdx.x & 31, n = l & 15, hi = l >> 4;
    v8f acc = {};
    for (int kb = 0; kb < LATD / 32; ++kb)
        acc = WMMA(load_tile(btA, (size_t)(b * 2 + mt) * 16 + kb),
                   load_tile(wbB, (size_t)kb * 256 + nt), acc);
    const float E = 2.71828182845904523536f;
#pragma unroll
    for (int r = 0; r < 8; ++r) {
        int m = mt * 16 + r + hi * 8;               // 0..31
        float s = acc[r] + b_bt[b * NF_ + m];
        int lane2, j2; pack_pos(n, m, lane2, j2);
        waveB[(((size_t)(b * 2 + 0) * 256 + nt) * 32 + lane2) * 16 + j2] = (h16)(E * cosf(s));
        waveB[(((size_t)(b * 2 + 1) * 256 + nt) * 32 + lane2) * 16 + j2] = (h16)(E * sinf(s));
    }
}

// ---- per-call: means -> wave_c scalars -> folded A1 packed (64x24->K32, MT=4) ----
__global__ void k_mean_wc_a1(const float* __restrict__ out, const h16* __restrict__ wd_c2w,
                             const float* __restrict__ b_c2w, const h16* __restrict__ wd_w2c,
                             h16* __restrict__ a1A) {
    int b = blockIdx.x;
    int t = threadIdx.x;
    __shared__ float red[256];
    __shared__ float meanS[NF_];
    __shared__ float wcS[CWCH];
    for (int ch = 0; ch < NF_; ++ch) {
        const float* p = out + ((size_t)b * NF_ + ch) * HW;
        float s = 0.f;
        for (int i = t; i < HW; i += 256) s += p[i];
        red[t] = s; __syncthreads();
        for (int w = 128; w > 0; w >>= 1) { if (t < w) red[t] += red[t + w]; __syncthreads(); }
        if (t == 0) meanS[ch] = red[0] * (1.0f / HW);
        __syncthreads();
    }
    if (t < CWCH) {
        float acc = b_c2w[b * CWCH + t];
        const h16* w = wd_c2w + (size_t)b * CWCH * NF_ + (size_t)t * NF_;
        for (int i = 0; i < NF_; ++i) acc += (float)w[i] * meanS[i];
        wcS[t] = acc;
    }
    __syncthreads();
    for (int idx = t; idx < 64 * 32; idx += 256) {
        int o = idx >> 5, k = idx & 31;
        float v = (k < CWCH) ? (float)wd_w2c[(size_t)b * 5632 + o * 88 + k] * wcS[k] : 0.0f;
        int lane, j; pack_pos(o & 15, k, lane, j);
        a1A[(((size_t)(b * 4 + (o >> 4))) * 32 + lane) * 16 + j] = (h16)v;
    }
}

// ---- depthwise [ident, gx, gy] -> g packed-B (96 ch, KB=3) ----
__global__ void k_grads(const float* __restrict__ out, h16* __restrict__ gB) {
    int bf = blockIdx.x; int b = bf >> 5; int f = bf & 31;
    __shared__ float tile[HW];
    const float* src = out + ((size_t)b * NF_ + f) * HW;
    for (int i = threadIdx.x; i < HW; i += blockDim.x) tile[i] = src[i];
    __syncthreads();
    for (int p = threadIdx.x; p < HW; p += blockDim.x) {
        int y = p >> 6, x = p & 63;
        auto at = [&](int yy, int xx) -> float {
            return ((unsigned)yy < 64u && (unsigned)xx < 64u) ? tile[yy * 64 + xx] : 0.0f;
        };
        float vals[3];
        vals[0] = tile[p];
        vals[1] = (at(y-1,x+1)+at(y,x+1)+at(y+1,x+1) - at(y-1,x-1)-at(y,x-1)-at(y+1,x-1)) * (1.0f/3.0f);
        vals[2] = (at(y+1,x-1)+at(y+1,x)+at(y+1,x+1) - at(y-1,x-1)-at(y-1,x)-at(y-1,x+1)) * (1.0f/3.0f);
        int nt = p >> 4, n = p & 15;
#pragma unroll
        for (int q = 0; q < 3; ++q) {
            int gch = 3 * f + q;
            int lane, j; pack_pos(n, gch & 31, lane, j);
            gB[(((size_t)(b * 3 + (gch >> 5)) * 256 + nt) * 32 + lane) * 16 + j] = (h16)vals[q];
        }
    }
}

// ---- h_pre = concat(grads_enc, wave_enc) -> packed-B (128 ch, KB=4) ----
__global__ void k_enc(const h16* __restrict__ gB, const h16* __restrict__ g2cA,
                      const float* __restrict__ b_g2c, const h16* __restrict__ a1A,
                      const h16* __restrict__ cwB, const h16* __restrict__ w2cA2,
                      const float* __restrict__ b_w2c, const h16* __restrict__ waveB,
                      h16* __restrict__ hpreB) {
    int b = blockIdx.y, nt = blockIdx.x;
    int mt = threadIdx.x >> 5;
    int l = threadIdx.x & 31, n = l & 15, hi = l >> 4;
    { // grads_enc
        v8f acc = {};
#pragma unroll
        for (int kb = 0; kb < 3; ++kb)
            acc = WMMA(load_tile(g2cA, (size_t)(b * 4 + mt) * 3 + kb),
                       load_tile(gB, (size_t)(b * 3 + kb) * 256 + nt), acc);
#pragma unroll
        for (int r = 0; r < 8; ++r) {
            int ch = mt * 16 + r + hi * 8;
            float v = acc[r] + b_g2c[b * 64 + ch];
            int lane2, j2; pack_pos(n, ch & 31, lane2, j2);
            hpreB[(((size_t)(b * 4 + (ch >> 5)) * 256 + nt) * 32 + lane2) * 16 + j2] = (h16)v;
        }
    }
    { // wave_enc
        v8f acc = {};
        acc = WMMA(load_tile(a1A, (size_t)(b * 4 + mt)), load_tile(cwB, (size_t)nt), acc);
#pragma unroll
        for (int kb = 0; kb < 2; ++kb)
            acc = WMMA(load_tile(w2cA2, (size_t)(b * 4 + mt) * 2 + kb),
                       load_tile(waveB, (size_t)(b * 2 + kb) * 256 + nt), acc);
#pragma unroll
        for (int r = 0; r < 8; ++r) {
            int ch = mt * 16 + r + hi * 8;
            float v = acc[r] + b_w2c[b * 64 + ch];
            int ch2 = 64 + ch;
            int lane2, j2; pack_pos(n, ch2 & 31, lane2, j2);
            hpreB[(((size_t)(b * 4 + (ch2 >> 5)) * 256 + nt) * 32 + lane2) * 16 + j2] = (h16)v;
        }
    }
}

// ---- instance-norm stats from packed hpre ----
__global__ void k_stats(const h16* __restrict__ hpreB, float* __restrict__ stats) {
    int idx = blockIdx.x;                  // b*128 + ch
    int b = idx >> 7, ch = idx & 127;
    int kb = ch >> 5, k = ch & 31;
    int hi2 = (k >> 3) & 1;
    int kk = k - 8 * hi2;
    int j = (kk < 8) ? kk : kk - 8;
    __shared__ float rs[256], rq[256];
    float s = 0.f, q = 0.f;
    for (int i = threadIdx.x; i < HW; i += 256) {
        int nt = i >> 4, n = i & 15;
        float v = (float)hpreB[(((size_t)(b * 4 + kb) * 256 + nt) * 32 + n + 16 * hi2) * 16 + j];
        s += v; q += v * v;
    }
    rs[threadIdx.x] = s; rq[threadIdx.x] = q; __syncthreads();
    for (int w = 128; w > 0; w >>= 1) {
        if (threadIdx.x < (unsigned)w) { rs[threadIdx.x] += rs[threadIdx.x + w]; rq[threadIdx.x] += rq[threadIdx.x + w]; }
        __syncthreads();
    }
    if (threadIdx.x == 0) {
        float m = rs[0] * (1.0f / HW);
        float var = rq[0] * (1.0f / HW) - m * m;
        stats[idx * 2]     = m;
        stats[idx * 2 + 1] = rsqrtf(var + 1e-5f);
    }
}

// ---- fold instance-norm scale into c1/sc packed A-matrices ----
__global__ void k_fold_w(const h16* __restrict__ rm_c1, const h16* __restrict__ rm_sc,
                         const float* __restrict__ stats,
                         h16* __restrict__ c1A, h16* __restrict__ scA) {
    int e = blockIdx.x * 256 + threadIdx.x;        // 16*96*128
    int b = e / (96 * 128);
    int rem = e % (96 * 128);
    int o = rem / 128, k = rem % 128;
    float rs = stats[(b * 128 + k) * 2 + 1];
    int lane, j;
    if (o < 64) {
        float v = (float)rm_c1[(size_t)b * 8192 + o * 128 + k] * rs;
        pack_pos(o & 15, k & 31, lane, j);
        c1A[(((size_t)(b * 4 + (o >> 4)) * 4 + (k >> 5)) * 32 + lane) * 16 + j] = (h16)v;
    } else {
        int oo = o - 64;
        float v = (float)rm_sc[(size_t)b * 4096 + oo * 128 + k] * rs;
        pack_pos(oo & 15, k & 31, lane, j);
        scA[(((size_t)(b * 2 + (oo >> 4)) * 4 + (k >> 5)) * 32 + lane) * 16 + j] = (h16)v;
    }
}

// ---- fold instance-norm shift into c1/sc biases ----
__global__ void k_fold_b(const h16* __restrict__ rm_c1, const h16* __restrict__ rm_sc,
                         const float* __restrict__ stats,
                         const float* __restrict__ b_c1, const float* __restrict__ b_sc,
                         float* __restrict__ bf_c1, float* __restrict__ bf_sc) {
    int t = blockIdx.x * blockDim.x + threadIdx.x;
    if (t >= B_ * 96) return;
    int b = t / 96, o = t % 96;
    float acc = 0.f;
    if (o < 64) {
        for (int k = 0; k < 128; ++k)
            acc += (float)rm_c1[(size_t)b * 8192 + o * 128 + k] * stats[(b * 128 + k) * 2 + 1] * stats[(b * 128 + k) * 2];
        bf_c1[b * 64 + o] = b_c1[b * 64 + o] - acc;
    } else {
        int oo = o - 64;
        for (int k = 0; k < 128; ++k)
            acc += (float)rm_sc[(size_t)b * 4096 + oo * 128 + k] * stats[(b * 128 + k) * 2 + 1] * stats[(b * 128 + k) * 2];
        bf_sc[b * 32 + oo] = b_sc[b * 32 + oo] - acc;
    }
}

// ---- residual block with register-cached B panel ----
__global__ void k_resblock(const h16* __restrict__ hpreB,
                           const h16* __restrict__ c1A, const float* __restrict__ bf_c1,
                           const h16* __restrict__ scA, const float* __restrict__ bf_sc,
                           const h16* __restrict__ c2A, const float* __restrict__ b_c2,
                           const float* __restrict__ leakp,
                           const float* __restrict__ out_prev, float* __restrict__ out_next) {
    int b = blockIdx.y, nt = blockIdx.x;
    int l = threadIdx.x & 31, n = l & 15, hi = l >> 4;
    __shared__ h16 r1pk[2 * 32 * 16] __attribute__((aligned(32)));   // 64-ch B panel, KB=2
    // cache the 4 hpre B-tiles once; reused by all 6 output M-tiles
    v16h hb[4];
#pragma unroll
    for (int kb = 0; kb < 4; ++kb)
        hb[kb] = load_tile(hpreB, (size_t)(b * 4 + kb) * 256 + nt);
#pragma unroll
    for (int t = 0; t < 4; ++t) {
        v8f acc = {};
#pragma unroll
        for (int kb = 0; kb < 4; ++kb)
            acc = WMMA(load_tile(c1A, (size_t)(b * 4 + t) * 4 + kb), hb[kb], acc);
#pragma unroll
        for (int r = 0; r < 8; ++r) {
            int ch = t * 16 + r + hi * 8;
            float v = fmaxf(acc[r] + bf_c1[b * 64 + ch], 0.0f);
            int lane2, j2; pack_pos(n, ch & 31, lane2, j2);
            r1pk[((size_t)((ch >> 5) * 32 + lane2)) * 16 + j2] = (h16)v;
        }
    }
    __syncthreads();
    float leak = fminf(fmaxf(leakp[0], 1e-3f), 1e3f);
    v16h rb[2];
#pragma unroll
    for (int kb = 0; kb < 2; ++kb)
        rb[kb] = *(const v16h*)(r1pk + ((size_t)kb * 32 + l) * 16);
#pragma unroll
    for (int t = 0; t < 2; ++t) {
        v8f accS = {};
#pragma unroll
        for (int kb = 0; kb < 4; ++kb)
            accS = WMMA(load_tile(scA, (size_t)(b * 2 + t) * 4 + kb), hb[kb], accS);
        v8f accC = {};
#pragma unroll
        for (int kb = 0; kb < 2; ++kb)
            accC = WMMA(load_tile(c2A, (size_t)(b * 2 + t) * 2 + kb), rb[kb], accC);
#pragma unroll
        for (int r = 0; r < 8; ++r) {
            int ch = t * 16 + r + hi * 8;
            float nv = accS[r] + bf_sc[b * 32 + ch] + accC[r] + b_c2[b * 32 + ch];
            size_t o = ((size_t)b * NF_ + ch) * HW + nt * 16 + n;
            out_next[o] = out_prev[o] + leak * nv;
        }
    }
}

// ---- final 1x1 conv + clip ----
__global__ void k_output(const float* __restrict__ out4, const float* __restrict__ ocw,
                         const float* __restrict__ ocb, float* __restrict__ out_clip,
                         float* __restrict__ out_raw) {
    int t = blockIdx.x * blockDim.x + threadIdx.x;
    if (t >= B_ * 3 * HW) return;
    int p = t % HW;
    int o = (t / HW) % 3;
    int b = t / (3 * HW);
    float acc = ocb[o];
    for (int c = 0; c < NF_; ++c)
        acc += ocw[o * NF_ + c] * out4[((size_t)b * NF_ + c) * HW + p];
    out_raw[t]  = acc;
    out_clip[t] = fminf(fmaxf(acc, -1.0f), 1.0f);
}

extern "C" void kernel_launch(void* const* d_in, const int* in_sizes, int n_in,
                              void* d_out, int out_size, void* d_ws, size_t ws_size,
                              hipStream_t stream) {
    (void)in_sizes; (void)n_in; (void)out_size; (void)ws_size;
    const float* lat       = (const float*)d_in[0];
    const float* seed      = (const float*)d_in[1];
    const float* leak      = (const float*)d_in[2];
    const float* wave_bias = (const float*)d_in[3];
    const float* cell_wave = (const float*)d_in[4];
    const float* bt_w  = (const float*)d_in[5];  const float* bt_b  = (const float*)d_in[6];
    const float* c2w_w = (const float*)d_in[7];  const float* c2w_b = (const float*)d_in[8];
    const float* w2c_w = (const float*)d_in[9];  const float* w2c_b = (const float*)d_in[10];
    const float* g2c_w = (const float*)d_in[11]; const float* g2c_b = (const float*)d_in[12];
    const float* sc_w  = (const float*)d_in[13]; const float* sc_b  = (const float*)d_in[14];
    const float* c1_w  = (const float*)d_in[15]; const float* c1_b  = (const float*)d_in[16];
    const float* c2_w  = (const float*)d_in[17]; const float* c2_b  = (const float*)d_in[18];
    const float* oc_w  = (const float*)d_in[19]; const float* oc_b  = (const float*)d_in[20];
    float* out = (float*)d_out;

    char* ws = (char*)d_ws;
    size_t off = 0;
    auto take = [&](size_t bytes) -> char* {
        char* p = ws + off;
        off += (bytes + 255) & ~(size_t)255;
        return p;
    };
    h16* latA     = (h16*)take((size_t)16 * 512 * 2);
    h16* wbB      = (h16*)take((size_t)16 * 256 * 512 * 2);
    h16* cwB      = (h16*)take((size_t)4 * 256 * 512 * 2);
    h16* btA      = (h16*)take((size_t)B_ * 2 * 16 * 512 * 2);
    h16* g2cA     = (h16*)take((size_t)B_ * 4 * 3 * 512 * 2);
    h16* c2A      = (h16*)take((size_t)B_ * 2 * 2 * 512 * 2);
    h16* c2w_rm   = (h16*)take((size_t)B_ * 768 * 2);
    h16* w2c_rm   = (h16*)take((size_t)B_ * 5632 * 2);
    h16* c1_rm    = (h16*)take((size_t)B_ * 8192 * 2);
    h16* sc_rm    = (h16*)take((size_t)B_ * 4096 * 2);
    h16* w2cA2    = (h16*)take((size_t)B_ * 4 * 2 * 512 * 2);
    h16* c1Af     = (h16*)take((size_t)B_ * 4 * 4 * 512 * 2);
    h16* scAf     = (h16*)take((size_t)B_ * 2 * 4 * 512 * 2);
    h16* a1A      = (h16*)take((size_t)B_ * 4 * 1 * 512 * 2);
    float* bb_bt  = (float*)take(B_ * 32 * 4);
    float* bb_c2w = (float*)take(B_ * 24 * 4);
    float* bb_w2c = (float*)take(B_ * 64 * 4);
    float* bb_g2c = (float*)take(B_ * 64 * 4);
    float* bb_sc  = (float*)take(B_ * 32 * 4);
    float* bb_c1  = (float*)take(B_ * 64 * 4);
    float* bb_c2  = (float*)take(B_ * 32 * 4);
    float* bf_c1  = (float*)take(B_ * 64 * 4);
    float* bf_sc  = (float*)take(B_ * 32 * 4);
    h16* waveB    = (h16*)take((size_t)B_ * 2 * 256 * 512 * 2);
    h16* gB       = (h16*)take((size_t)B_ * 3 * 256 * 512 * 2);
    h16* hpreB    = (h16*)take((size_t)B_ * 4 * 256 * 512 * 2);
    float* stats  = (float*)take(B_ * 128 * 2 * 4);

    // one-time packs
    k_pack_lat<<<(16 * 512) / 256, 256, 0, stream>>>(lat, latA);
    k_pack_wb<<<(512 * HW) / 256, 256, 0, stream>>>(wave_bias, wbB);
    k_pack_cw<<<(4 * 32 * HW) / 256, 256, 0, stream>>>(cell_wave, cwB);

    // dynamic weights
    k_wd_gemm_pk<<<16384 / 16, 32, 0, stream>>>(latA, bt_w,  btA,  32, 512);
    k_wd_gemm_pk<<< 6144 / 16, 32, 0, stream>>>(latA, g2c_w, g2cA, 64, 96);
    k_wd_gemm_pk<<< 2048 / 16, 32, 0, stream>>>(latA, c2_w,  c2A,  32, 64);
    k_wd_gemm_rm<<<  768 / 16, 32, 0, stream>>>(latA, c2w_w, c2w_rm,  768);
    k_wd_gemm_rm<<< 5632 / 16, 32, 0, stream>>>(latA, w2c_w, w2c_rm, 5632);
    k_wd_gemm_rm<<< 8192 / 16, 32, 0, stream>>>(latA, c1_w,  c1_rm,  8192);
    k_wd_gemm_rm<<< 4096 / 16, 32, 0, stream>>>(latA, sc_w,  sc_rm,  4096);
    k_pack_w2cA2<<<(B_ * 64 * 64) / 256, 256, 0, stream>>>(w2c_rm, w2cA2);

    // dynamic biases (WMMA)
    k_bias_gemm<<<2, 32, 0, stream>>>(latA, bt_b,  bb_bt,  32);
    k_bias_gemm<<<2, 32, 0, stream>>>(latA, c2w_b, bb_c2w, 24);
    k_bias_gemm<<<4, 32, 0, stream>>>(latA, w2c_b, bb_w2c, 64);
    k_bias_gemm<<<4, 32, 0, stream>>>(latA, g2c_b, bb_g2c, 64);
    k_bias_gemm<<<2, 32, 0, stream>>>(latA, sc_b,  bb_sc,  32);
    k_bias_gemm<<<4, 32, 0, stream>>>(latA, c1_b,  bb_c1,  64);
    k_bias_gemm<<<2, 32, 0, stream>>>(latA, c2_b,  bb_c2,  32);

    float* emb0 = out + 196608;
    k_seed<<<(B_ * NF_ * HW) / 256, 256, 0, stream>>>(seed, emb0);
    k_surface<<<dim3(256, B_), 64, 0, stream>>>(btA, bb_bt, wbB, waveB);

    for (int c = 0; c < 4; ++c) {
        const float* embc = out + 196608 + (size_t)c * (B_ * NF_ * HW);
        float* embn       = out + 196608 + (size_t)(c + 1) * (B_ * NF_ * HW);
        k_mean_wc_a1<<<B_, 256, 0, stream>>>(embc, c2w_rm, bb_c2w, w2c_rm, a1A);
        k_grads<<<B_ * NF_, 256, 0, stream>>>(embc, gB);
        k_enc<<<dim3(256, B_), 128, 0, stream>>>(gB, g2cA, bb_g2c, a1A,
                                                 cwB + (size_t)c * 256 * 512,
                                                 w2cA2, bb_w2c, waveB, hpreB);
        k_stats<<<B_ * 128, 256, 0, stream>>>(hpreB, stats);
        k_fold_w<<<(B_ * 96 * 128) / 256, 256, 0, stream>>>(c1_rm, sc_rm, stats, c1Af, scAf);
        k_fold_b<<<(B_ * 96 + 255) / 256, 256, 0, stream>>>(c1_rm, sc_rm, stats, bb_c1, bb_sc, bf_c1, bf_sc);
        k_resblock<<<dim3(256, B_), 32, 0, stream>>>(hpreB, c1Af, bf_c1, scAf, bf_sc,
                                                     c2A, bb_c2, leak, embc, embn);
    }
    const float* emb4 = out + 196608 + (size_t)4 * (B_ * NF_ * HW);
    k_output<<<(B_ * 3 * HW) / 256, 256, 0, stream>>>(emb4, oc_w, oc_b,
                                                      out, out + 196608 + (size_t)5 * (B_ * NF_ * HW));
}
